// AttentionWithPairBias_52338471469580
// MI455X (gfx1250) — matrix operand order repaired and checked by reading, not compiled, not run on hardware
//
#include <hip/hip_runtime.h>
#include <math.h>

typedef __attribute__((ext_vector_type(16))) _Float16 v16h;
typedef __attribute__((ext_vector_type(8)))  _Float16 v8h;
typedef __attribute__((ext_vector_type(8)))  float    v8f;
typedef __attribute__((ext_vector_type(4)))  int      v4i;

#define AS1 __attribute__((address_space(1)))
#define AS3 __attribute__((address_space(3)))

#if defined(__has_builtin)
#if __has_builtin(__builtin_amdgcn_global_load_async_to_lds_b128)
#define HAVE_ASYNC_LDS 1
#endif
#endif

__device__ __forceinline__ v8f wmma_f16(v16h a, v16h b, v8f c) {
  // D = A(16x32 f16) * B(32x16 f16) + C(16x16 f32)
  return __builtin_amdgcn_wmma_f32_16x16x32_f16(false, a, false, b, (short)0, c, false, false);
}

// Assemble an A-style fragment from two contiguous 8-half (16B) runs.
__device__ __forceinline__ v16h ld2x8(const _Float16* p0, const _Float16* p1) {
  union { v16h v; v8h h[2]; } u;
  u.h[0] = *(const v8h*)p0;
  u.h[1] = *(const v8h*)p1;
  return u.v;
}

__device__ __forceinline__ float fast_rcp(float x) {
#if __has_builtin(__builtin_amdgcn_rcpf)
  return __builtin_amdgcn_rcpf(x);
#else
  return 1.f / x;
#endif
}

#ifdef HAVE_ASYNC_LDS
__device__ __forceinline__ void async_copy16(const _Float16* g, _Float16* l) {
  __builtin_amdgcn_global_load_async_to_lds_b128(
      (AS1 v4i*)g, (AS3 v4i*)l, 0, 0);
}
__device__ __forceinline__ void async_wait0() {
#if __has_builtin(__builtin_amdgcn_s_wait_asynccnt)
  __builtin_amdgcn_s_wait_asynccnt(0);
#else
  asm volatile("s_wait_asynccnt 0" ::: "memory");
#endif
}
#endif

// B=2, N=512, CM=256, CZ=128, H=8, KD=32
#define NPOS (2*512*512)
#define AS_LD 264   // padded LDS stride (halves) for 16x256 A tiles
#define LG_LD 516   // padded stride (floats) for logits rows
#define WT_LD 520   // padded stride (halves) for weight rows
#define LS_LD 136   // padded stride (halves) for 128-wide f16 rows

// ---------------------------------------------------------------------------
// K0: one-time weight conversion to f16, stored K-CONTIGUOUS (transposed):
// W4t[n][k] n in [0,1024) = concat(query*KD^-0.5, key, value, gating); k in [0,256)
// Wot[n][k] = output_w[k][n]
__global__ void k_prep(const float* __restrict__ qw, const float* __restrict__ kw,
                       const float* __restrict__ vw, const float* __restrict__ gw,
                       const float* __restrict__ ow,
                       _Float16* __restrict__ W4t, _Float16* __restrict__ Wot) {
  int i = blockIdx.x * 256 + threadIdx.x;
  if (i < 1024 * 256) {
    int n = i >> 8, k = i & 255, c = n & 255;
    float v;
    if (n < 256)      v = qw[k * 256 + c] * 0.17677669529663687f; // 1/sqrt(32)
    else if (n < 512) v = kw[k * 256 + c];
    else if (n < 768) v = vw[k * 256 + c];
    else              v = gw[k * 256 + c];
    W4t[i] = (_Float16)v;
  }
  if (i < 256 * 256) {
    int n = i >> 8, k = i & 255;
    Wot[i] = (_Float16)ow[k * 256 + n];
  }
}

// ---------------------------------------------------------------------------
// K1: LayerNorm over msa_act rows (256 wide), emit f16
__global__ void k_msa_ln(const float* __restrict__ msa, const float* __restrict__ sc,
                         const float* __restrict__ off, _Float16* __restrict__ out) {
  __shared__ float rs[8], rs2[8];
  int row = blockIdx.x, t = threadIdx.x;
  int wv = t >> 5, lane = t & 31;
  float x = msa[row * 256 + t];
  float s = x, s2 = x * x;
#pragma unroll
  for (int m = 16; m >= 1; m >>= 1) { s += __shfl_xor(s, m, 32); s2 += __shfl_xor(s2, m, 32); }
  if (lane == 0) { rs[wv] = s; rs2[wv] = s2; }
  __syncthreads();
  if (t < 32) {
    float a  = (lane < 8) ? rs[lane]  : 0.f;
    float a2 = (lane < 8) ? rs2[lane] : 0.f;
#pragma unroll
    for (int m = 4; m >= 1; m >>= 1) { a += __shfl_xor(a, m, 32); a2 += __shfl_xor(a2, m, 32); }
    if (lane == 0) { rs[0] = a; rs2[0] = a2; }
  }
  __syncthreads();
  float mean = rs[0] * (1.f / 256.f);
  float var  = rs2[0] * (1.f / 256.f) - mean * mean;
  float inv  = rsqrtf(var + 1e-5f);
  out[row * 256 + t] = (_Float16)((x - mean) * inv * sc[t] + off[t]);
}

// ---------------------------------------------------------------------------
// K2: fused q/k/v/gate projection GEMM [1024x256] x [256x1024] via WMMA.
// q/k row-major [bq][h*32+c]; v transposed to vT[b][h*32+c][k-seq].
__global__ void k_qkvg(const _Float16* __restrict__ X, const _Float16* __restrict__ W4t,
                       const float* __restrict__ gating_b,
                       _Float16* __restrict__ qF, _Float16* __restrict__ kF,
                       _Float16* __restrict__ vT, float* __restrict__ gateF) {
  __shared__ _Float16 As[16 * AS_LD];
  int t = threadIdx.x, lane = t & 31;
  int m0 = blockIdx.x * 16;
  int n0 = blockIdx.y * 128 + (t >> 5) * 16;
#ifdef HAVE_ASYNC_LDS
  for (int cc = t; cc < 512; cc += 256) {
    int row = cc >> 5, col = (cc & 31) * 8;
    async_copy16(X + (m0 + row) * 256 + col, As + row * AS_LD + col);
  }
  async_wait0();
#else
  for (int i = t; i < 16 * 256; i += 256)
    As[(i >> 8) * AS_LD + (i & 255)] = X[(m0 + (i >> 8)) * 256 + (i & 255)];
#endif
  __syncthreads();
  v8f acc = {};
  int am = lane & 15, akh = 16 * ((lane >> 4) & 1);
  int bn = n0 + (lane & 15), bkh = 16 * ((lane >> 4) & 1);
  for (int kk = 0; kk < 8; kk++) {
    int k0 = kk * 32;
    v16h a = ld2x8(As + am * AS_LD + k0 + akh, As + am * AS_LD + k0 + 8 + akh);
    v16h b = *(const v16h*)(W4t + (size_t)bn * 256 + k0 + bkh);
    acc = wmma_f16(a, b, acc);
  }
  int c = bn & 255;
#pragma unroll
  for (int r = 0; r < 8; r++) {
    int m = m0 + r + 8 * ((lane >> 4) & 1);
    float v = acc[r];
    if (bn < 256)      qF[m * 256 + c] = (_Float16)v;            // already scaled
    else if (bn < 512) kF[m * 256 + c] = (_Float16)v;
    else if (bn < 768) vT[((size_t)(m >> 9) * 256 + c) * 512 + (m & 511)] = (_Float16)v;
    else               gateF[m * 256 + c] = fast_rcp(1.f + __expf(-(v + gating_b[c])));
  }
}

// ---------------------------------------------------------------------------
// K3: single pass over pair_act: LayerNorm(128) + 128->8 head-bias projection
// one wave per (b,q,k) position; bias layout [b][h][q][k]
__global__ void k_pairbias(const float* __restrict__ pair, const float* __restrict__ mask,
                           const float* __restrict__ sc, const float* __restrict__ off,
                           const float* __restrict__ f2w, float* __restrict__ biasO) {
  __shared__ float F2t[8 * 128];   // transposed: F2t[h][c]
  __shared__ float S[128], O[128];
  int t = threadIdx.x;
  for (int i = t; i < 1024; i += 256) F2t[i] = f2w[(i & 127) * 8 + (i >> 7)];
  if (t < 128) { S[t] = sc[t]; O[t] = off[t]; }
  __syncthreads();
  int wv = t >> 5, lane = t & 31;
  int pos = blockIdx.x * 8 + wv;                 // (b*512+q)*512+k
  int b = pos >> 18, rem = pos & 262143;
  int q = rem >> 9, k = rem & 511;
  const float4* p4 = (const float4*)(pair + (size_t)pos * 128);
  float4 x = p4[lane];
  if (pos + 8 < NPOS)
    __builtin_prefetch((const float4*)(pair + (size_t)(pos + 8) * 128) + lane, 0, 1);
  float s = x.x + x.y + x.z + x.w;
  float s2 = x.x * x.x + x.y * x.y + x.z * x.z + x.w * x.w;
#pragma unroll
  for (int m = 16; m >= 1; m >>= 1) { s += __shfl_xor(s, m, 32); s2 += __shfl_xor(s2, m, 32); }
  float mean = s * (1.f / 128.f);
  float var  = s2 * (1.f / 128.f) - mean * mean;
  float inv  = rsqrtf(var + 1e-5f);
  int c0 = lane * 4;
  float n0 = (x.x - mean) * inv * S[c0 + 0] + O[c0 + 0];
  float n1 = (x.y - mean) * inv * S[c0 + 1] + O[c0 + 1];
  float n2 = (x.z - mean) * inv * S[c0 + 2] + O[c0 + 2];
  float n3 = (x.w - mean) * inv * S[c0 + 3] + O[c0 + 3];
  float myb = 0.f;
#pragma unroll
  for (int h = 0; h < 8; h++) {
    const float* Fh = F2t + h * 128 + c0;
    float p = n0 * Fh[0] + n1 * Fh[1] + n2 * Fh[2] + n3 * Fh[3];
#pragma unroll
    for (int m = 16; m >= 1; m >>= 1) p += __shfl_xor(p, m, 32);
    if (lane == h) myb = p;
  }
  if (lane < 8)
    biasO[(size_t)((b * 8 + lane) * 512 + q) * 512 + k] = myb + 1e9f * (mask[b * 512 + k] - 1.f);
}

// ---------------------------------------------------------------------------
// K4: attention core per (b,h,16-row q tile). Emits logits_update (pre-bias)
// and gated weighted_avg (f16, layout [bq][h*32+c]).
__global__ void k_attn(const _Float16* __restrict__ qF, const _Float16* __restrict__ kF,
                       const _Float16* __restrict__ vT, const float* __restrict__ gateF,
                       const float* __restrict__ biasG, float* __restrict__ logitsU,
                       _Float16* __restrict__ wavg) {
  __shared__ float    Lg[16 * LG_LD];   // logits+bias
  __shared__ _Float16 Wt[16 * WT_LD];   // softmax weights (f16)
  __shared__ float    Part[4][32][9];   // K-split partials (padded)
  int t = threadIdx.x, wv = t >> 5, lane = t & 31;
  int qt = blockIdx.x, h = blockIdx.y, b = blockIdx.z;
  int q0 = qt * 16, bbase = b * 512;
  size_t lbase = (size_t)((b * 8 + h) * 512) * 512;
  int am = lane & 15, akh = 16 * ((lane >> 4) & 1);
  int bkh = 16 * ((lane >> 4) & 1);

  // A fragment: queries, K = KD = 32 (one WMMA K-step)
  const _Float16* qrow = qF + (bbase + q0 + am) * 256 + h * 32;
  v16h afrag = ld2x8(qrow + akh, qrow + 8 + akh);

  // Phase 1: logits tiles (QK^T), stream logits_update, stage logits+bias in LDS
  for (int kt = wv; kt < 32; kt += 4) {
    int kb = kt * 16;
    // B: keys, contiguous 16 halves per lane (32B)
    v16h bfrag = *(const v16h*)(kF + (size_t)(bbase + kb + (lane & 15)) * 256 + h * 32 + bkh);
    v8f z = {};
    v8f d = wmma_f16(afrag, bfrag, z);
    int kc = kb + (lane & 15);
#pragma unroll
    for (int r = 0; r < 8; r++) {
      int qr = r + 8 * ((lane >> 4) & 1);
      float lu = d[r];
      logitsU[lbase + (size_t)(q0 + qr) * 512 + kc] = lu;
      Lg[qr * LG_LD + kc] = lu + biasG[lbase + (size_t)(q0 + qr) * 512 + kc];
    }
  }
  __syncthreads();

  // Phase 2: softmax over k (512), 4 rows per wave
  for (int rr = 0; rr < 4; rr++) {
    int row = wv * 4 + rr;
    float vals[16];
    float mx = -3.0e38f;
#pragma unroll
    for (int i = 0; i < 16; i++) { vals[i] = Lg[row * LG_LD + lane + 32 * i]; mx = fmaxf(mx, vals[i]); }
#pragma unroll
    for (int m = 16; m >= 1; m >>= 1) mx = fmaxf(mx, __shfl_xor(mx, m, 32));
    float sum = 0.f;
#pragma unroll
    for (int i = 0; i < 16; i++) { vals[i] = __expf(vals[i] - mx); sum += vals[i]; }
#pragma unroll
    for (int m = 16; m >= 1; m >>= 1) sum += __shfl_xor(sum, m, 32);
    float inv = fast_rcp(sum);
#pragma unroll
    for (int i = 0; i < 16; i++) Wt[row * WT_LD + lane + 32 * i] = (_Float16)(vals[i] * inv);
  }
  __syncthreads();

  // Phase 3: weights * V. waves (0,1)->c[0:16), (2,3)->c[16:32); K split in half
  int ntile = wv >> 1, khalf = wv & 1;
  int c0 = ntile * 16;
  const _Float16* vbase = vT + ((size_t)b * 256 + h * 32 + c0 + (lane & 15)) * 512;
  v8f acc = {};
  for (int kc8 = 0; kc8 < 8; kc8++) {
    int k0 = khalf * 256 + kc8 * 32;
    v16h a = ld2x8(Wt + am * WT_LD + k0 + akh, Wt + am * WT_LD + k0 + 8 + akh);
    v16h bfr = *(const v16h*)(vbase + k0 + bkh);   // contiguous over K
    acc = wmma_f16(a, bfr, acc);
  }
#pragma unroll
  for (int r = 0; r < 8; r++) Part[wv][lane][r] = acc[r];
  __syncthreads();
  if (khalf == 0) {
#pragma unroll
    for (int r = 0; r < 8; r++) {
      float v = Part[wv][lane][r] + Part[wv + 1][lane][r];
      int qr = r + 8 * ((lane >> 4) & 1);
      int m = bbase + q0 + qr;
      int cc = h * 32 + c0 + (lane & 15);
      wavg[m * 256 + cc] = (_Float16)(v * gateF[m * 256 + cc]);
    }
  }
}

// ---------------------------------------------------------------------------
// K5: output projection [1024x256] x [256x256] + bias -> msa_out (f32)
__global__ void k_outproj(const _Float16* __restrict__ Xg, const _Float16* __restrict__ Wot,
                          const float* __restrict__ ob, float* __restrict__ out) {
  __shared__ _Float16 As[16 * AS_LD];
  int t = threadIdx.x, lane = t & 31;
  int m0 = blockIdx.x * 16;
  int n0 = blockIdx.y * 128 + (t >> 5) * 16;
#ifdef HAVE_ASYNC_LDS
  for (int cc = t; cc < 512; cc += 256) {
    int row = cc >> 5, col = (cc & 31) * 8;
    async_copy16(Xg + (m0 + row) * 256 + col, As + row * AS_LD + col);
  }
  async_wait0();
#else
  for (int i = t; i < 16 * 256; i += 256)
    As[(i >> 8) * AS_LD + (i & 255)] = Xg[(m0 + (i >> 8)) * 256 + (i & 255)];
#endif
  __syncthreads();
  v8f acc = {};
  int am = lane & 15, akh = 16 * ((lane >> 4) & 1);
  int bn = n0 + (lane & 15), bkh = 16 * ((lane >> 4) & 1);
  for (int kk = 0; kk < 8; kk++) {
    int k0 = kk * 32;
    v16h a = ld2x8(As + am * AS_LD + k0 + akh, As + am * AS_LD + k0 + 8 + akh);
    v16h b = *(const v16h*)(Wot + (size_t)bn * 256 + k0 + bkh);
    acc = wmma_f16(a, b, acc);
  }
#pragma unroll
  for (int r = 0; r < 8; r++) {
    int m = m0 + r + 8 * ((lane >> 4) & 1);
    out[m * 256 + bn] = acc[r] + ob[bn];
  }
}

// ---------------------------------------------------------------------------
// K6: pair update. Per (b,q): relu(logits^T(bqkh) @ upd0 + b0) -> [16x128] f16,
// then [16x128] x [128x128] WMMA GEMM + b1 -> pair_out, looped over 32 k-tiles.
// upd1 weights staged transposed in LDS: U1t[n][k].
__global__ void k_pairupd(const float* __restrict__ logitsU,
                          const float* __restrict__ u0w, const float* __restrict__ u0b,
                          const float* __restrict__ u1w, const float* __restrict__ u1b,
                          float* __restrict__ pairO) {
  __shared__ _Float16 U1t[128 * LS_LD];  // transposed upd1 weights
  __shared__ float    U0[8 * 128];
  __shared__ float    B0[128], B1[128];
  __shared__ float    Lt[16][8];
  __shared__ _Float16 Ls[16 * LS_LD];
  int t = threadIdx.x, wv = t >> 5, lane = t & 31;
  for (int i = t; i < 128 * 128; i += 256) {
    int d = i >> 7, n = i & 127;
    U1t[n * LS_LD + d] = (_Float16)u1w[i];
  }
  for (int i = t; i < 1024; i += 256) U0[i] = u0w[i];
  if (t < 128) { B0[t] = u0b[t]; B1[t] = u1b[t]; }
  __syncthreads();
  int bq = blockIdx.x;
  int b = bq >> 9, q = bq & 511;
  size_t lbase = (size_t)(b * 8) * 262144 + (size_t)q * 512; // + h*262144 + k
  size_t obase = (size_t)bq * 512 * 128;
  int am = lane & 15, akh = 16 * ((lane >> 4) & 1);
  int n0 = wv * 16, bn = n0 + (lane & 15), bkh = 16 * ((lane >> 4) & 1);
  for (int kt = 0; kt < 32; kt++) {
    int k0 = kt * 16;
    if (t < 128) {
      int kk = t >> 3, h = t & 7;
      Lt[kk][h] = logitsU[lbase + (size_t)h * 262144 + k0 + kk];
    }
    __syncthreads();
    for (int i = t; i < 2048; i += 256) {
      int kk = i >> 7, d = i & 127;
      float a = B0[d];
#pragma unroll
      for (int h = 0; h < 8; h++) a += Lt[kk][h] * U0[h * 128 + d];
      Ls[kk * LS_LD + d] = (_Float16)fmaxf(a, 0.f);
    }
    __syncthreads();
    v8f acc = {};
#pragma unroll
    for (int kc = 0; kc < 4; kc++) {
      int kb = kc * 32;
      v16h a  = ld2x8(Ls + am * LS_LD + kb + akh, Ls + am * LS_LD + kb + 8 + akh);
      v16h bb = ld2x8(U1t + bn * LS_LD + kb + bkh, U1t + bn * LS_LD + kb + bkh + 8);
      acc = wmma_f16(a, bb, acc);
    }
#pragma unroll
    for (int r = 0; r < 8; r++) {
      int kk = r + 8 * ((lane >> 4) & 1);
      pairO[obase + (size_t)(k0 + kk) * 128 + bn] = acc[r] + B1[bn];
    }
    __syncthreads();
  }
}

// ---------------------------------------------------------------------------
extern "C" void kernel_launch(void* const* d_in, const int* in_sizes, int n_in,
                              void* d_out, int out_size, void* d_ws, size_t ws_size,
                              hipStream_t stream) {
  (void)in_sizes; (void)n_in; (void)out_size; (void)ws_size;
  const float* msa_act  = (const float*)d_in[0];
  const float* msa_mask = (const float*)d_in[1];
  const float* pair_act = (const float*)d_in[2];
  const float* qn_scale = (const float*)d_in[3];
  const float* qn_off   = (const float*)d_in[4];
  const float* pn_scale = (const float*)d_in[5];
  const float* pn_off   = (const float*)d_in[6];
  const float* f2w      = (const float*)d_in[7];
  const float* qw       = (const float*)d_in[8];
  const float* kw       = (const float*)d_in[9];
  const float* vw       = (const float*)d_in[10];
  const float* gw       = (const float*)d_in[11];
  const float* gb       = (const float*)d_in[12];
  const float* ow       = (const float*)d_in[13];
  const float* ob       = (const float*)d_in[14];
  const float* u0w      = (const float*)d_in[15];
  const float* u0b      = (const float*)d_in[16];
  const float* u1w      = (const float*)d_in[17];
  const float* u1b      = (const float*)d_in[18];

  char* w = (char*)d_ws;
  size_t o = 0;
  auto alloc = [&](size_t bytes) { void* p = w + o; o = (o + bytes + 255) & ~(size_t)255; return p; };
  _Float16* msa16 = (_Float16*)alloc((size_t)1024 * 256 * 2);
  _Float16* W4t   = (_Float16*)alloc((size_t)1024 * 256 * 2);
  _Float16* Wot   = (_Float16*)alloc((size_t)256 * 256 * 2);
  _Float16* qF    = (_Float16*)alloc((size_t)1024 * 256 * 2);
  _Float16* kF    = (_Float16*)alloc((size_t)1024 * 256 * 2);
  _Float16* vT    = (_Float16*)alloc((size_t)1024 * 256 * 2);
  float*    gateF = (float*)alloc((size_t)1024 * 256 * 4);
  float*    biasF = (float*)alloc((size_t)2 * 8 * 512 * 512 * 4);
  float*    logU  = (float*)alloc((size_t)2 * 8 * 512 * 512 * 4);
  _Float16* wavg  = (_Float16*)alloc((size_t)1024 * 256 * 2);

  float* msa_out  = (float*)d_out;
  float* pair_out = msa_out + 2 * 512 * 256;

  hipLaunchKernelGGL(k_prep,     dim3(1024),      dim3(256), 0, stream, qw, kw, vw, gw, ow, W4t, Wot);
  hipLaunchKernelGGL(k_msa_ln,   dim3(1024),      dim3(256), 0, stream, msa_act, qn_scale, qn_off, msa16);
  hipLaunchKernelGGL(k_qkvg,     dim3(64, 8),     dim3(256), 0, stream, msa16, W4t, gb, qF, kF, vT, gateF);
  hipLaunchKernelGGL(k_pairbias, dim3(65536),     dim3(256), 0, stream, pair_act, msa_mask, pn_scale, pn_off, f2w, biasF);
  hipLaunchKernelGGL(k_attn,     dim3(32, 8, 2),  dim3(128), 0, stream, qF, kF, vT, gateF, biasF, logU, wavg);
  hipLaunchKernelGGL(k_outproj,  dim3(64, 2),     dim3(256), 0, stream, wavg, Wot, ob, msa_out);
  hipLaunchKernelGGL(k_pairupd,  dim3(1024),      dim3(256), 0, stream, logU, u0w, u0b, u1w, u1b, pair_out);
}
